// MultiAttentionOld_48361331753100
// MI455X (gfx1250) — compile-verified, matrix operand order checked by reading, not run on hardware
//
#include <hip/hip_runtime.h>
#include <hip/hip_bf16.h>

// ---------------------------------------------------------------------------
// MultiHeadAttention for MI455X (gfx1250, wave32, WMMA 16x16x32 f16)
//   B=4, S=2048, D=1024, H=16, K=64
// Pipeline: qkv projections (WMMA GEMM, f32->f16; V written transposed)
//   -> flash attention (WMMA, online softmax, double-buffered LDS tiles
//      staged with GLOBAL_LOAD_ASYNC_TO_LDS_B128)
//   -> output projection (WMMA GEMM, f16 x f16 -> f32).
// ---------------------------------------------------------------------------

typedef __attribute__((ext_vector_type(16))) _Float16 v16h;
typedef __attribute__((ext_vector_type(8)))  _Float16 v8h;
typedef __attribute__((ext_vector_type(8)))  float    v8f;
typedef int v4i __attribute__((vector_size(16)));   // matches builtin's V4i

union F8  { v8f v;  float f[8]; };
union H16 { v16h v; _Float16 h[16]; unsigned int u[8]; };
union H8  { v8h v;  _Float16 h[8]; };

#define SEQ   2048
#define DIM   1024
#define NH    16
#define HD    64

#if __has_builtin(__builtin_amdgcn_global_load_async_to_lds_b128) && \
    __has_builtin(__builtin_amdgcn_s_wait_asynccnt)
#define USE_ASYNC 1
#else
#define USE_ASYNC 0
#endif

// 16B chunk copy global -> LDS; async (ASYNCcnt-tracked DMA) when available.
__device__ __forceinline__ void tile_chunk_copy(const _Float16* g, _Float16* l) {
#if USE_ASYNC
  __builtin_amdgcn_global_load_async_to_lds_b128(
      (__attribute__((address_space(1))) v4i*)(void*)(const_cast<_Float16*>(g)),
      (__attribute__((address_space(3))) v4i*)(void*)l, 0, 0);
#else
  *reinterpret_cast<uint4*>(l) = *reinterpret_cast<const uint4*>(g);
#endif
}

__device__ __forceinline__ void tiles_ready() {
#if USE_ASYNC
  __builtin_amdgcn_s_wait_asynccnt(0);
#endif
  __syncthreads();
}

// ---------------------------------------------------------------------------
// A/B 16x32 f16 fragment from a row-major tile (rows = M for A / N for B,
// cols = k).  CDNA5 layout: lanes 0-15 hold row (lane), k in {0..7,16..23};
// lanes 16-31 hold the same row, k in {8..15,24..31}.
// ---------------------------------------------------------------------------
__device__ __forceinline__ v16h load_frag16x32(const _Float16* base, int ld, int lane) {
  const int r  = lane & 15;
  const int kb = (lane >> 4) << 3;     // 0 or 8
  const unsigned int* p0 = reinterpret_cast<const unsigned int*>(base + r * ld + kb);
  const unsigned int* p1 = reinterpret_cast<const unsigned int*>(base + r * ld + kb + 16);
  H16 f;
  f.u[0] = p0[0]; f.u[1] = p0[1]; f.u[2] = p0[2]; f.u[3] = p0[3];
  f.u[4] = p1[0]; f.u[5] = p1[1]; f.u[6] = p1[2]; f.u[7] = p1[3];
  return f.v;
}

__device__ __forceinline__ v8f wmma_f16(v16h a, v16h b, v8f c) {
  return __builtin_amdgcn_wmma_f32_16x16x32_f16(false, a, false, b, (short)0, c,
                                                false, false);
}

// ---------------------------------------------------------------------------
// Tiled GEMM:  C[M=8192, N=1024] = A[M,1024] * W[1024, N]
//  OUTMODE 0: A f32, W per-head [H,D,64], out f16 [B,H,S,64]   (Q, K proj)
//  OUTMODE 1: A f32, W per-head [H,D,64], out f16 [B,H,64,S]   (V proj, ^T)
//  OUTMODE 2: A f16, W flat [1024,1024],  out f32 [M,1024]     (final proj)
// Workgroup: 256 threads / 8 waves; tile 128(M) x 64(N); k-step 32.
// ---------------------------------------------------------------------------
template <int OUTMODE>
__global__ __launch_bounds__(256) void gemm_kernel(const void* __restrict__ Ain,
                                                   const float* __restrict__ W,
                                                   void* __restrict__ Out) {
  constexpr bool PROJ = (OUTMODE != 2);
  constexpr int MT = 128, KT = 32, LDA = 40, LDW = 40;
  __shared__ _Float16 Xs[MT * LDA];   // 128 rows x 32 k (padded)
  __shared__ _Float16 Ws[64 * LDW];   // 64 n  x 32 k (W transposed, padded)

  const int tid  = threadIdx.x;
  const int wave = tid >> 5;
  const int lane = tid & 31;
  const int m0   = blockIdx.x * MT;
  const int n0   = blockIdx.y * 64;

  const float* Wt  = PROJ ? (W + (size_t)(n0 >> 6) * DIM * HD) : (W + n0);
  const int    wrs = PROJ ? HD : DIM;

  F8 acc[4];
#pragma unroll
  for (int c = 0; c < 4; ++c)
#pragma unroll
    for (int r = 0; r < 8; ++r) acc[c].f[r] = 0.0f;

  for (int d0 = 0; d0 < DIM; d0 += KT) {
    __syncthreads();
    // --- stage A tile (convert f32 -> f16 when PROJ) ---
    {
      const int row = tid >> 1;
      const int j0  = (tid & 1) * 16;
      if (PROJ) {
        const float* ar = (const float*)Ain + (size_t)(m0 + row) * DIM + d0 + j0;
#pragma unroll
        for (int j = 0; j < 16; ++j) Xs[row * LDA + j0 + j] = (_Float16)ar[j];
      } else {
        const _Float16* ar = (const _Float16*)Ain + (size_t)(m0 + row) * DIM + d0 + j0;
        *reinterpret_cast<uint4*>(&Xs[row * LDA + j0]) =
            *reinterpret_cast<const uint4*>(ar);
        *reinterpret_cast<uint4*>(&Xs[row * LDA + j0 + 8]) =
            *reinterpret_cast<const uint4*>(ar + 8);
      }
    }
    // --- stage W tile transposed: Ws[n][k] = W[d0+k][n0+n] ---
    {
      const int dd  = tid >> 3;           // 0..31
      const int kf0 = (tid & 7) * 8;      // 0..56
      const float* wr = Wt + (size_t)(d0 + dd) * wrs + kf0;
#pragma unroll
      for (int j = 0; j < 8; ++j) Ws[(kf0 + j) * LDW + dd] = (_Float16)wr[j];
    }
    __syncthreads();

    const v16h a = load_frag16x32(&Xs[wave * 16 * LDA], LDA, lane);
#pragma unroll
    for (int c = 0; c < 4; ++c) {
      const v16h b = load_frag16x32(&Ws[c * 16 * LDW], LDW, lane);
      acc[c].v = wmma_f16(a, b, acc[c].v);
    }
  }

  // --- store: C layout lane = n + 16*(mrel>=8), vgpr = mrel%8 ---
  const int ncol = lane & 15;
  const int mup  = (lane >> 4) * 8;
#pragma unroll
  for (int c = 0; c < 4; ++c) {
    const int n = n0 + c * 16 + ncol;
#pragma unroll
    for (int r = 0; r < 8; ++r) {
      const int row = m0 + wave * 16 + mup + r;
      if (OUTMODE == 2) {
        ((float*)Out)[(size_t)row * DIM + n] = acc[c].f[r];
      } else {
        const int b = row >> 11, s = row & (SEQ - 1);
        const int h = n >> 6, kf = n & (HD - 1);
        const size_t idx =
            (OUTMODE == 0)
                ? ((((size_t)b * NH + h) * SEQ) + s) * HD + kf   // [B,H,S,64]
                : ((((size_t)b * NH + h) * HD) + kf) * SEQ + s;  // [B,H,64,S]
        ((_Float16*)Out)[idx] = (_Float16)acc[c].f[r];
      }
    }
  }
}

// ---------------------------------------------------------------------------
// Flash attention.  grid = (S/128, B*H), 256 threads (8 waves), each wave
// owns 16 query rows.  kv blocks of 64, double-buffered in LDS and staged
// with async global->LDS copies.  V arrives pre-transposed [B,H,64,S] so both
// tiles are plain row-segment copies.  Computes S^T = K*Q^T so softmax rows
// are lane-local and the S^T accumulator converts lane-locally into the
// B-fragment of P^T for O^T += V^T * P^T.
// ---------------------------------------------------------------------------
#define KVB 64
#define LDK 72   // padded row stride (16B-aligned rows, conflict-free frags)

__global__ __launch_bounds__(256) void flash_attn_kernel(
    const _Float16* __restrict__ qg, const _Float16* __restrict__ kg,
    const _Float16* __restrict__ vtg, _Float16* __restrict__ cat) {
  __shared__ _Float16 Ks[2][KVB * LDK];   // [buf][kv][d]
  __shared__ _Float16 Vts[2][HD * LDK];   // [buf][d_out][kv]

  const int bh   = blockIdx.y;                 // b*16 + h
  const int tid  = threadIdx.x;
  const int wave = tid >> 5;
  const int lane = tid & 31;
  const int q0   = blockIdx.x * 128 + wave * 16;

  const _Float16* Qb  = qg  + (size_t)bh * SEQ * HD;
  const _Float16* Kb  = kg  + (size_t)bh * SEQ * HD;
  const _Float16* VbT = vtg + (size_t)bh * HD * SEQ;   // [64][S]

  // Q^T B-fragments: lane holds query row (q0 + lane%16), d striped per ISA.
  const v16h qt0 = load_frag16x32(Qb + (size_t)q0 * HD, HD, lane);
  const v16h qt1 = load_frag16x32(Qb + (size_t)q0 * HD + 32, HD, lane);

  F8 o[4];
#pragma unroll
  for (int f = 0; f < 4; ++f)
#pragma unroll
    for (int r = 0; r < 8; ++r) o[f].f[r] = 0.0f;
  float mrun = -1e30f, lrun = 0.0f;

  // Stage one 64x64 tile pair into LDS buffer p (2 b128 chunks per tile/thread).
  auto stage = [&](int p, int kv0) {
#pragma unroll
    for (int i = 0; i < 2; ++i) {
      const int c = tid * 2 + i;          // 512 chunks of 8 halves
      const int row = c >> 3, c8 = c & 7;
      tile_chunk_copy(Kb + (size_t)(kv0 + row) * HD + c8 * 8,
                      &Ks[p][row * LDK + c8 * 8]);
      tile_chunk_copy(VbT + (size_t)row * SEQ + kv0 + c8 * 8,
                      &Vts[p][row * LDK + c8 * 8]);
    }
  };

  stage(0, 0);
  tiles_ready();

  for (int kv0 = 0; kv0 < SEQ; kv0 += KVB) {
    const int p = (kv0 >> 6) & 1;
    const bool more = (kv0 + KVB) < SEQ;
    if (more) stage(1 - p, kv0 + KVB);   // overlap next copy with this compute

    // --- S^T = K * Q^T  (4 chunks of 16 kv, k-dim = 64 -> 2 WMMAs each) ---
    F8 s[4];
#pragma unroll
    for (int c = 0; c < 4; ++c) {
      const v16h ka0 = load_frag16x32(&Ks[p][c * 16 * LDK], LDK, lane);
      const v16h ka1 = load_frag16x32(&Ks[p][c * 16 * LDK + 32], LDK, lane);
      v8f a = {};
      a = wmma_f16(ka0, qt0, a);
      a = wmma_f16(ka1, qt1, a);
      s[c].v = a;
    }
    // --- online softmax (scale 1/sqrt(64) = 0.125) ---
    float tm = -1e30f;
#pragma unroll
    for (int c = 0; c < 4; ++c)
#pragma unroll
      for (int r = 0; r < 8; ++r) {
        s[c].f[r] *= 0.125f;
        tm = fmaxf(tm, s[c].f[r]);
      }
    tm = fmaxf(tm, __shfl_xor(tm, 16, 32));
    const float mnew = fmaxf(mrun, tm);
    const float corr = __expf(mrun - mnew);
    float rs = 0.0f;
#pragma unroll
    for (int c = 0; c < 4; ++c)
#pragma unroll
      for (int r = 0; r < 8; ++r) {
        const float e = __expf(s[c].f[r] - mnew);
        s[c].f[r] = e;
        rs += e;
      }
    rs += __shfl_xor(rs, 16, 32);
    lrun = lrun * corr + rs;
    mrun = mnew;
#pragma unroll
    for (int f = 0; f < 4; ++f)
#pragma unroll
      for (int r = 0; r < 8; ++r) o[f].f[r] *= corr;

    // --- lane-local S^T(C-layout) -> P^T(B-fragment) conversion ---
    H16 p0, p1;
#pragma unroll
    for (int h2 = 0; h2 < 8; ++h2) {
      p0.h[h2]     = (_Float16)s[0].f[h2];
      p0.h[h2 + 8] = (_Float16)s[1].f[h2];
      p1.h[h2]     = (_Float16)s[2].f[h2];
      p1.h[h2 + 8] = (_Float16)s[3].f[h2];
    }
    // --- O^T += V^T * P^T ---
#pragma unroll
    for (int f = 0; f < 4; ++f) {
      const v16h va0 = load_frag16x32(&Vts[p][f * 16 * LDK], LDK, lane);
      const v16h va1 = load_frag16x32(&Vts[p][f * 16 * LDK + 32], LDK, lane);
      o[f].v = wmma_f16(va0, p0.v, o[f].v);
      o[f].v = wmma_f16(va1, p1.v, o[f].v);
    }

    if (more) tiles_ready();             // next buffer resident for all waves
  }

  // --- epilogue: normalize and write concat[B,S,H*64] as f16 ---
  const float inv = 1.0f / lrun;
  const int m   = lane & 15;
  const int kup = (lane >> 4) * 8;
  const int b = bh >> 4, h = bh & 15;
  _Float16* dst = cat + ((size_t)b * SEQ + q0 + m) * DIM + h * HD;
#pragma unroll
  for (int f = 0; f < 4; ++f) {
    H8 ov;
#pragma unroll
    for (int r = 0; r < 8; ++r) ov.h[r] = (_Float16)(o[f].f[r] * inv);
    *reinterpret_cast<v8h*>(dst + f * 16 + kup) = ov.v;
  }
}

// ---------------------------------------------------------------------------
extern "C" void kernel_launch(void* const* d_in, const int* in_sizes, int n_in,
                              void* d_out, int out_size, void* d_ws,
                              size_t ws_size, hipStream_t stream) {
  (void)in_sizes; (void)n_in; (void)out_size; (void)ws_size;
  const float* query = (const float*)d_in[0];
  const float* key   = (const float*)d_in[1];
  const float* value = (const float*)d_in[2];
  const float* Wq    = (const float*)d_in[3];
  const float* Wk    = (const float*)d_in[4];
  const float* Wv    = (const float*)d_in[5];
  const float* Wo    = (const float*)d_in[6];

  const size_t HEADS_ELEMS = (size_t)4 * NH * SEQ * HD;  // 8,388,608
  _Float16* qb  = (_Float16*)d_ws;
  _Float16* kb  = qb + HEADS_ELEMS;
  _Float16* vbt = kb + HEADS_ELEMS;   // V stored transposed [B,H,64,S]
  _Float16* cat = vbt + HEADS_ELEMS;

  const dim3 blk(256);
  gemm_kernel<0><<<dim3(64, 16), blk, 0, stream>>>(query, Wq, qb);
  gemm_kernel<0><<<dim3(64, 16), blk, 0, stream>>>(key,   Wk, kb);
  gemm_kernel<1><<<dim3(64, 16), blk, 0, stream>>>(value, Wv, vbt);
  flash_attn_kernel<<<dim3(16, 64), blk, 0, stream>>>(qb, kb, vbt, cat);
  gemm_kernel<2><<<dim3(64, 16), blk, 0, stream>>>(cat, Wo, (float*)d_out);
}
